// GAT_55843164783128
// MI455X (gfx1250) — compile-verified, matrix operand order checked by reading, not run on hardware
//
#include <hip/hip_runtime.h>

// ---------------------------------------------------------------------------
// GAT (2-layer) for MI455X / gfx1250.
// GEMM1 runs through v_wmma_f32_16x16x32_bf16 (wave32, 16x16 tile per wave).
// Attention/aggregation is atomic/scatter bound and stays in fp32 VALU+atomics.
// ---------------------------------------------------------------------------

typedef __attribute__((ext_vector_type(16))) __bf16 v16bf;
typedef __attribute__((ext_vector_type(8)))  float  v8f;

union ABFrag { uint4 q[2]; v16bf v; };

#define NEG_SLOPE 0.2f

// fp32 -> bf16 bits, round-to-nearest-even
__device__ __forceinline__ unsigned short f2bf(float f) {
  unsigned u = __float_as_uint(f);
  unsigned r = u + 0x7FFFu + ((u >> 16) & 1u);
  return (unsigned short)(r >> 16);
}

// monotonic float<->uint encoding so atomicMax(u32) == float max
__device__ __forceinline__ unsigned fenc(float f) {
  unsigned u = __float_as_uint(f);
  return (u & 0x80000000u) ? ~u : (u | 0x80000000u);
}
__device__ __forceinline__ float fdec(unsigned u) {
  u = (u & 0x80000000u) ? (u & 0x7FFFFFFFu) : ~u;
  return __uint_as_float(u);
}

// ---- conversions -----------------------------------------------------------
__global__ void cvt_x_bf16(const float* __restrict__ x, unsigned short* __restrict__ xb, int n) {
  int i = blockIdx.x * blockDim.x + threadIdx.x;
  if (i < n) xb[i] = f2bf(x[i]);
}

// w1t[col][k] = bf16(W1[k][col]); W1 is [128][128] row-major
__global__ void cvt_w1t_bf16(const float* __restrict__ W1, unsigned short* __restrict__ w1t) {
  int i = blockIdx.x * blockDim.x + threadIdx.x;   // i < 16384
  if (i >= 128 * 128) return;
  int col = i >> 7, k = i & 127;
  w1t[i] = f2bf(W1[k * 128 + col]);
}

// ---- GEMM1: h1 = x @ W1 via WMMA bf16 -> f32 -------------------------------
// block = 256 threads = 8 waves; block b owns rows [16b,16b+16);
// wave w owns cols [16w,16w+16); K=128 in 4 steps of 32.
__global__ void gemm1_wmma(const unsigned short* __restrict__ xb,
                           const unsigned short* __restrict__ w1t,
                           float* __restrict__ h1, int N) {
  int wave = threadIdx.x >> 5;
  int lane = threadIdx.x & 31;
  int rbase = blockIdx.x * 16;
  int row = rbase + (lane & 15);
  if (row >= N) row = N - 1;                    // clamp (N is 16-divisible anyway)
  int col = wave * 16 + (lane & 15);
  int kb  = (lane >> 4) * 8;                    // K half-split per ISA A/B layout

  v8f acc = {};
  #pragma unroll
  for (int k0 = 0; k0 < 128; k0 += 32) {
    ABFrag a, b;
    const uint4* ap = (const uint4*)(xb  + (size_t)row * 128 + k0 + kb);
    const uint4* bp = (const uint4*)(w1t + (size_t)col * 128 + k0 + kb);
    a.q[0] = ap[0];  a.q[1] = ap[2];            // K {+0..7} and {+16..23}
    b.q[0] = bp[0];  b.q[1] = bp[2];
    acc = __builtin_amdgcn_wmma_f32_16x16x32_bf16(
        false, a.v, false, b.v, (short)0, acc, false, false);
  }

  int nidx = lane & 15;
  int mtop = (lane >> 4) * 8;                   // C/D: vgpr r -> M = mtop + r
  float* outp = h1 + (size_t)(rbase + mtop) * 128 + wave * 16 + nidx;
  if (rbase + 16 <= N) {
    // full tile: branch-free epilogue (common case; N % 16 == 0 here)
    #pragma unroll
    for (int r = 0; r < 8; ++r) outp[(size_t)r * 128] = acc[r];
  } else {
    #pragma unroll
    for (int r = 0; r < 8; ++r)
      if (rbase + mtop + r < N) outp[(size_t)r * 128] = acc[r];
  }
}

// ---- edge prep: narrow int64 indices to int32, append self loops ---------------------
__global__ void prep_edges(const long long* __restrict__ ei, int E, int N,
                           int* __restrict__ src32, int* __restrict__ dst32) {
  int e = blockIdx.x * blockDim.x + threadIdx.x;
  int EE = E + N;
  if (e >= EE) return;
  int s, d;
  if (e < E) { s = (int)ei[e]; d = (int)ei[(size_t)E + e]; }
  else       { s = d = e - E; }
  src32[e] = s; dst32[e] = d;
}

// ---- per-node attention halves (layer 1) -----------------------------------
__global__ void node_scores(const float* __restrict__ h,
                            const float* __restrict__ as, const float* __restrict__ ad,
                            float* __restrict__ ss, float* __restrict__ sd, int N) {
  int n = blockIdx.x * blockDim.x + threadIdx.x;
  if (n >= N) return;
  const float* hp = h + (size_t)n * 128;
  float accs = 0.f, accd = 0.f;
  #pragma unroll 8
  for (int k = 0; k < 128; ++k) { float v = hp[k]; accs += v * as[k]; accd += v * ad[k]; }
  ss[n] = accs; sd[n] = accd;
}

// ---- init: out = bias, m = enc sentinel, z = 0 -----------------------------
__global__ void init_nf(float* __restrict__ out, const float* __restrict__ b, int Fdim,
                        unsigned* __restrict__ m, float* __restrict__ z, int N) {
  int i = blockIdx.x * blockDim.x + threadIdx.x;
  if (i < N * Fdim) out[i] = b[i % Fdim];
  if (i < N) { m[i] = 0u; z[i] = 0.f; }
}

// ---- edge pass 1: leaky score + atomic segment max -------------------------
__global__ void edge_scores(const float* __restrict__ ss, const float* __restrict__ sd,
                            const int* __restrict__ src32, const int* __restrict__ dst32,
                            float* __restrict__ eraw, unsigned* __restrict__ m, int EE) {
  int e = blockIdx.x * blockDim.x + threadIdx.x;
  if (e >= EE) return;
  int s = src32[e], d = dst32[e];
  float v = ss[s] + sd[d];
  v = (v >= 0.f) ? v : NEG_SLOPE * v;
  eraw[e] = v;
  atomicMax(&m[d], fenc(v));
}

// ---- edge pass 2: exp(e - m) + atomic segment sum --------------------------
__global__ void edge_exp(float* __restrict__ eraw, const unsigned* __restrict__ m,
                         const int* __restrict__ dst32, float* __restrict__ z, int EE) {
  int e = blockIdx.x * blockDim.x + threadIdx.x;
  if (e >= EE) return;
  int d = dst32[e];
  float ex = __expf(eraw[e] - fdec(m[d]));
  eraw[e] = ex;
  atomicAdd(&z[d], ex);
}

// ---- weighted scatter, F=128: one wave per edge ----------------------------
__global__ void gat_aggregate128(const float* __restrict__ h, const float* __restrict__ ex,
                                 const float* __restrict__ z,
                                 const int* __restrict__ src32, const int* __restrict__ dst32,
                                 float* __restrict__ out, int EE) {
  int e = blockIdx.x * 8 + (threadIdx.x >> 5);
  if (e >= EE) return;
  int lane = threadIdx.x & 31;
  int s = src32[e], d = dst32[e];
  float alpha = ex[e] / z[d];
  float4 m4 = ((const float4*)(h + (size_t)s * 128))[lane];
  float* op = out + (size_t)d * 128 + lane * 4;
  atomicAdd(op + 0, alpha * m4.x);
  atomicAdd(op + 1, alpha * m4.y);
  atomicAdd(op + 2, alpha * m4.z);
  atomicAdd(op + 3, alpha * m4.w);
}

// ---- ReLU + layer-2 projection (F=2) + layer-2 attention halves ------------
__global__ void relu_proj2(const float* __restrict__ out1, const float* __restrict__ W2,
                           const float* __restrict__ as2, const float* __restrict__ ad2,
                           float* __restrict__ h2, float* __restrict__ ss,
                           float* __restrict__ sd, int N) {
  int n = blockIdx.x * blockDim.x + threadIdx.x;
  if (n >= N) return;
  const float* p = out1 + (size_t)n * 128;
  float a0 = 0.f, a1 = 0.f;
  #pragma unroll 8
  for (int k = 0; k < 128; ++k) {
    float r = p[k]; r = r > 0.f ? r : 0.f;
    a0 += r * W2[2 * k]; a1 += r * W2[2 * k + 1];
  }
  h2[2 * n] = a0; h2[2 * n + 1] = a1;
  ss[n] = a0 * as2[0] + a1 * as2[1];
  sd[n] = a0 * ad2[0] + a1 * ad2[1];
}

// ---- weighted scatter, F=2: one thread per edge ----------------------------
__global__ void gat_aggregate2(const float* __restrict__ h2, const float* __restrict__ ex,
                               const float* __restrict__ z,
                               const int* __restrict__ src32, const int* __restrict__ dst32,
                               float* __restrict__ out, int EE) {
  int e = blockIdx.x * blockDim.x + threadIdx.x;
  if (e >= EE) return;
  int s = src32[e], d = dst32[e];
  float alpha = ex[e] / z[d];
  atomicAdd(&out[2 * d + 0], alpha * h2[2 * s + 0]);
  atomicAdd(&out[2 * d + 1], alpha * h2[2 * s + 1]);
}

// ---------------------------------------------------------------------------
extern "C" void kernel_launch(void* const* d_in, const int* in_sizes, int n_in,
                              void* d_out, int out_size, void* d_ws, size_t ws_size,
                              hipStream_t stream) {
  const float*     x      = (const float*)d_in[0];
  const long long* ei     = (const long long*)d_in[1];   // int64 edge_index [2,E]
  const float*     W1     = (const float*)d_in[2];
  const float*     a_src1 = (const float*)d_in[3];
  const float*     a_dst1 = (const float*)d_in[4];
  const float*     b1     = (const float*)d_in[5];
  const float*     W2     = (const float*)d_in[6];
  const float*     a_src2 = (const float*)d_in[7];
  const float*     a_dst2 = (const float*)d_in[8];
  const float*     b2     = (const float*)d_in[9];
  float*           out    = (float*)d_out;
  (void)n_in; (void)out_size; (void)ws_size;

  const int F  = 128;
  const int N  = in_sizes[0] / F;
  const int E  = in_sizes[1] / 2;
  const int EE = E + N;

  // workspace carve-out (256B aligned)
  char* ws = (char*)d_ws; size_t off = 0;
  auto carve = [&](size_t bytes) -> char* {
    char* p = ws + off; off += (bytes + 255) & ~(size_t)255; return p;
  };
  unsigned short* xb    = (unsigned short*)carve((size_t)N * F * 2);
  unsigned short* w1t   = (unsigned short*)carve((size_t)F * F * 2);
  float*          h1    = (float*)carve((size_t)N * F * 4);
  float*          out1  = (float*)carve((size_t)N * F * 4);
  float*          s_s   = (float*)carve((size_t)N * 4);
  float*          s_d   = (float*)carve((size_t)N * 4);
  float*          h2    = (float*)carve((size_t)N * 2 * 4);
  unsigned*       mseg  = (unsigned*)carve((size_t)N * 4);
  float*          zseg  = (float*)carve((size_t)N * 4);
  int*            src32 = (int*)carve((size_t)EE * 4);
  int*            dst32 = (int*)carve((size_t)EE * 4);
  float*          eraw  = (float*)carve((size_t)EE * 4);

  const int T = 256;
  auto g = [&](long long n) -> int { return (int)((n + T - 1) / T); };

  // layer 1: projection via WMMA
  cvt_x_bf16 <<<g((long long)N * F), T, 0, stream>>>(x, xb, N * F);
  cvt_w1t_bf16<<<g(F * F), T, 0, stream>>>(W1, w1t);
  gemm1_wmma <<<(N + 15) / 16, T, 0, stream>>>(xb, w1t, h1, N);

  // edges (shared by both layers)
  prep_edges<<<g(EE), T, 0, stream>>>(ei, E, N, src32, dst32);

  // layer 1 attention + aggregation
  node_scores<<<g(N), T, 0, stream>>>(h1, a_src1, a_dst1, s_s, s_d, N);
  init_nf    <<<g((long long)N * F), T, 0, stream>>>(out1, b1, F, mseg, zseg, N);
  edge_scores<<<g(EE), T, 0, stream>>>(s_s, s_d, src32, dst32, eraw, mseg, EE);
  edge_exp   <<<g(EE), T, 0, stream>>>(eraw, mseg, dst32, zseg, EE);
  gat_aggregate128<<<(EE + 7) / 8, T, 0, stream>>>(h1, eraw, zseg, src32, dst32, out1, EE);

  // layer 2: ReLU + tiny projection + attention + aggregation
  relu_proj2 <<<g(N), T, 0, stream>>>(out1, W2, a_src2, a_dst2, h2, s_s, s_d, N);
  init_nf    <<<g((long long)N * 2), T, 0, stream>>>(out, b2, 2, mseg, zseg, N);
  edge_scores<<<g(EE), T, 0, stream>>>(s_s, s_d, src32, dst32, eraw, mseg, EE);
  edge_exp   <<<g(EE), T, 0, stream>>>(eraw, mseg, dst32, zseg, EE);
  gat_aggregate2<<<g(EE), T, 0, stream>>>(h2, eraw, zseg, src32, dst32, out, EE);
}
